// BaseNeighborNN_20925080666532
// MI455X (gfx1250) — matrix operand order repaired and checked by reading, not compiled, not run on hardware
//
#include <hip/hip_runtime.h>

// Problem constants (from reference)
#define BB      16
#define NP      1024
#define KK      20
#define IN_DIM  78
#define INP     96          // padded feature dim (3 x 32)
#define HID     128
#define BOX     10.0f

// Fused kernel geometry
#define PPB       8                 // particles per block
#define ROWS_PB   (PPB * KK)        // 160 rows per block
#define THREADS_F ROWS_PB           // 160 threads = 5 wave32
#define WAVES_F   5

typedef __attribute__((ext_vector_type(16))) _Float16 v16h;
typedef __attribute__((ext_vector_type(8)))  float    v8f;
typedef __attribute__((ext_vector_type(4)))  int      v4i;

// Address-space qualified 16B vector types for the async global->LDS builtin
// (diagnostic showed the param is a pointer to a 4-int vector, AS-qualified).
typedef __attribute__((address_space(1))) v4i v4i_g;
typedef __attribute__((address_space(3))) v4i v4i_l;

// gfx1250 async global->LDS path (ASYNCcnt-tracked), guarded so a toolchain
// without the builtins falls back to plain cooperative loads.
#if defined(__has_builtin)
#if __has_builtin(__builtin_amdgcn_global_load_async_to_lds_b128) && \
    __has_builtin(__builtin_amdgcn_s_wait_asynccnt)
#define HAVE_ASYNC_LDS 1
#endif
#endif

// float atomic max via native integer LDS atomics (ds_max_i32 / ds_min_u32).
// Valid for IEEE floats: non-negative floats order as signed ints; negative
// floats order inversely as unsigned ints. Buffer initialized to -2.0f.
__device__ __forceinline__ void atomicMaxF(int* addr, float v) {
  if (v >= 0.0f) atomicMax(addr, __float_as_int(v));
  else           atomicMin((unsigned int*)addr, __float_as_uint(v));
}

// ---------------------------------------------------------------------------
// Kernel 1: repack weights into WMMA B-fragment order, f16, zero-padded.
// Fragment layout (per 16x16x32 step s, col-tile ct, lane, element e):
//   lane<16 : col = 16*ct + lane,      K = 32*s + {0..7} then 32*s + {16..23}
//   lane>=16: col = 16*ct + lane - 16, K = 32*s + {8..15} then 32*s + {24..31}
// Stored linearly: ((s*8 + ct)*32 + lane)*16 + e  -> one v16h per lane.
// ---------------------------------------------------------------------------
__global__ void prep_weights(const float* __restrict__ w1,
                             const float* __restrict__ w2,
                             _Float16* __restrict__ w1f,
                             _Float16* __restrict__ w2f) {
  for (int t = threadIdx.x; t < 3 * 8 * 32 * 16; t += blockDim.x) {
    int e = t & 15, lane = (t >> 4) & 31, ct = (t >> 9) & 7, s = t >> 12;
    int col = 16 * ct + (lane & 15);
    int k = 32 * s + (lane >> 4) * 8 + ((e < 8) ? e : (8 + e));
    float v = (k < IN_DIM) ? w1[k * HID + col] : 0.0f;
    w1f[t] = (_Float16)v;
  }
  for (int t = threadIdx.x; t < 4 * 8 * 32 * 16; t += blockDim.x) {
    int e = t & 15, lane = (t >> 4) & 31, ct = (t >> 9) & 7, s = t >> 12;
    int col = 16 * ct + (lane & 15);
    int k = 32 * s + (lane >> 4) * 8 + ((e < 8) ? e : (8 + e));
    w2f[t] = (_Float16)w2[k * HID + col];
  }
}

// ---------------------------------------------------------------------------
// Kernel 2: top-K=20 nearest neighbors (min-image PBC), per batch.
// Grid: BB * (NP/256) blocks of 256. Positions for the batch staged into LDS
// via GLOBAL_LOAD_ASYNC_TO_LDS_B128 when available (ASYNCcnt path).
// Per-thread top-K by worst-slot replacement (order irrelevant: max-pool later).
// ---------------------------------------------------------------------------
__global__ void __launch_bounds__(256) topk_kernel(const float* __restrict__ pos,
                                                   int* __restrict__ nbr) {
  __shared__ float ps[NP * 3];          // 12 KB
  __shared__ float kd[256 * KK];        // 20 KB
  __shared__ int   ki[256 * KK];        // 20 KB
  const int b = blockIdx.x >> 2;
  const int i = ((blockIdx.x & 3) << 8) + threadIdx.x;
  const float* pb = pos + b * NP * 3;

#if defined(HAVE_ASYNC_LDS)
  {
    // 12 KB = 768 x 16B; 256 lanes x 3 iterations, each lane one b128 copy.
    v4i_g* gsrc = (v4i_g*)pb;           // C-style cast: strips const, sets AS(1)
    v4i_l* ldst = (v4i_l*)ps;           // AS(3)
#pragma unroll
    for (int t = 0; t < 3; ++t) {
      const int idx = threadIdx.x + t * 256;
      __builtin_amdgcn_global_load_async_to_lds_b128(gsrc + idx, ldst + idx, 0, 0);
    }
    __builtin_amdgcn_s_wait_asynccnt(0);
  }
#else
  for (int t = threadIdx.x; t < NP * 3; t += 256) ps[t] = pb[t];
#endif
  __syncthreads();

  const float xi = ps[i * 3 + 0], yi = ps[i * 3 + 1], zi = ps[i * 3 + 2];
  float* md = kd + threadIdx.x * KK;
  int*   mi = ki + threadIdx.x * KK;
  for (int t = 0; t < KK; ++t) { md[t] = 3.0e30f; mi[t] = 0; }
  float worst = 3.0e30f;
  int   ws = 0;
  const float invbox = 1.0f / BOX;
  for (int j = 0; j < NP; ++j) {
    float dx = xi - ps[j * 3 + 0];
    float dy = yi - ps[j * 3 + 1];
    float dz = zi - ps[j * 3 + 2];
    dx -= BOX * rintf(dx * invbox);
    dy -= BOX * rintf(dy * invbox);
    dz -= BOX * rintf(dz * invbox);
    float r2 = dx * dx + dy * dy + dz * dz;
    if (j == i) continue;                 // reference excludes self via +1e9
    if (r2 < worst) {
      md[ws] = r2; mi[ws] = j;
      worst = -1.0f;
      for (int t = 0; t < KK; ++t)
        if (md[t] > worst) { worst = md[t]; ws = t; }
    }
  }
  int* o = nbr + (b * NP + i) * KK;
  for (int t = 0; t < KK; ++t) o[t] = mi[t];
}

// ---------------------------------------------------------------------------
// Kernel 3: fused features + MLP (WMMA) + max-pool + output layer.
// Block: 160 threads = 5 wave32, handles 8 particles x 20 neighbors = 160 rows.
// Each lane builds one feature row (f16, padded to 96) in LDS; each wave runs
// two 16-row WMMA tiles through layer1 (3 k-steps x 8 col-tiles) and layer2
// (4 k-steps x 8 col-tiles) with f32 accumulation, tanh in f32, and pools via
// native LDS atomics. Weight fragments stream from L2 (global), prefetched.
// ---------------------------------------------------------------------------
__global__ void __launch_bounds__(THREADS_F) fused_mlp(
    const float* __restrict__ pos, const float* __restrict__ ori,
    const int* __restrict__ nbr,
    const _Float16* __restrict__ w1f, const float* __restrict__ b1,
    const _Float16* __restrict__ w2f, const float* __restrict__ b2,
    const float* __restrict__ w3, const float* __restrict__ b3,
    float* __restrict__ out) {
  __shared__ _Float16 featS[ROWS_PB * INP];   // 30720 B
  __shared__ _Float16 hS[WAVES_F][16 * HID];  // 20480 B
  __shared__ int pooledS[PPB * HID];          //  4096 B

  const int tid  = threadIdx.x;
  const int lane = tid & 31;
  const int w    = tid >> 5;
  const int rsel = lane & 15;
  const int hi   = lane >> 4;     // 0: lanes 0-15, 1: lanes 16-31

  // warm L2/L0 for the weight-fragment streams (global_prefetch_b8)
  __builtin_prefetch(w1f + lane * 64, 0, 3);
  __builtin_prefetch(w2f + lane * 64, 0, 3);

  for (int t = tid; t < PPB * HID; t += THREADS_F)
    pooledS[t] = __float_as_int(-2.0f);       // tanh >= -1 > -2

  // ---- feature row per lane (row index == tid) ----
  {
    const int lr = tid;
    const int p = lr / KK, k = lr % KK;
    const int g = blockIdx.x * PPB + p;
    const int b = g >> 10, n = g & (NP - 1);
    const int j = nbr[g * KK + k];
    const float* Pn = pos + (b * NP + n) * 3;
    const float* Pj = pos + (b * NP + j) * 3;
    const float* Rpg = ori + (b * NP + n) * 9;
    const float* Rng = ori + (b * NP + j) * 9;
    float rp[9], rn[9];
#pragma unroll
    for (int t = 0; t < 9; ++t) { rp[t] = Rpg[t]; rn[t] = Rng[t]; }

    float d0 = Pn[0] - Pj[0], d1 = Pn[1] - Pj[1], d2 = Pn[2] - Pj[2];
    d0 -= BOX * rintf(d0 * (1.0f / BOX));
    d1 -= BOX * rintf(d1 * (1.0f / BOX));
    d2 -= BOX * rintf(d2 * (1.0f / BOX));
    const float r2 = d0 * d0 + d1 * d1 + d2 * d2;
    const float r = sqrtf(r2), inv = 1.0f / r;
    const float u0 = d0 * inv, u1 = d1 * inv, u2 = d2 * inv;

    _Float16* f = featS + lr * INP;
    f[0] = (_Float16)r;  f[1] = (_Float16)inv;
    f[2] = (_Float16)u0; f[3] = (_Float16)u1; f[4] = (_Float16)u2;

#pragma unroll
    for (int a = 0; a < 3; ++a) {
#pragma unroll
      for (int l = 0; l < 3; ++l) {
        const float e0 = rp[3 * a + 0] * rn[3 * l + 0];
        const float e1 = rp[3 * a + 1] * rn[3 * l + 1];
        const float e2 = rp[3 * a + 2] * rn[3 * l + 2];
        f[5 + 3 * a + l] = (_Float16)(e0 + e1 + e2);                 // dot
        f[14 + 9 * a + 3 * l + 0] = (_Float16)e0;                    // elem
        f[14 + 9 * a + 3 * l + 1] = (_Float16)e1;
        f[14 + 9 * a + 3 * l + 2] = (_Float16)e2;
        f[41 + 3 * a + l] = (_Float16)sqrtf(e0 * e0 + e1 * e1 + e2 * e2);
        const float rl = rp[0 + a] * rn[0 + l] + rp[3 + a] * rn[3 + l] +
                         rp[6 + a] * rn[6 + l];                      // rel = Rp^T Rn
        f[62 + 3 * a + l] = (_Float16)rl;
      }
    }
#pragma unroll
    for (int a = 0; a < 3; ++a) {
      const float ax = rp[3 * a], ay = rp[3 * a + 1], az = rp[3 * a + 2];
      const float bx = rn[3 * a], by = rn[3 * a + 1], bz = rn[3 * a + 2];
      const float cx = ay * bz - az * by;
      const float cy = az * bx - ax * bz;
      const float cz = ax * by - ay * bx;
      f[50 + 3 * a + 0] = (_Float16)cx;                              // cross
      f[50 + 3 * a + 1] = (_Float16)cy;
      f[50 + 3 * a + 2] = (_Float16)cz;
      f[59 + a] = (_Float16)sqrtf(cx * cx + cy * cy + cz * cz);      // cross_norm
      const float dp = u0 * ax + u1 * ay + u2 * az;
      f[71 + a] = (_Float16)expf(-dp * dp);                          // rbf_p
      const float dn = u0 * bx + u1 * by + u2 * bz;
      f[74 + a] = (_Float16)expf(-dn * dn);                          // rbf_n
    }
    float tr = 0.0f;
#pragma unroll
    for (int t = 0; t < 9; ++t) tr += rp[t] * rn[t];                 // trace(Rp^T Rn)
    const float cl = fminf(fmaxf((tr - 1.0f) * 0.5f, -1.0f + 1e-6f), 1.0f - 1e-6f);
    f[77] = (_Float16)acosf(cl);
#pragma unroll
    for (int t = IN_DIM; t < INP; ++t) f[t] = (_Float16)0.0f;        // pad to 96
  }
  __syncthreads();

  const v16h* w1v = (const v16h*)w1f;
  const v16h* w2v = (const v16h*)w2f;
  const v8f zero = {};

#pragma unroll
  for (int tile = 0; tile < 2; ++tile) {
    const int rowbase = 32 * w + 16 * tile;

    // ---- layer 1: (16 x 96) x (96 x 128), 24 WMMAs ----
    v8f acc[8];
#pragma unroll
    for (int ct = 0; ct < 8; ++ct) acc[ct] = zero;
#pragma unroll
    for (int s = 0; s < 3; ++s) {
      // A-fragment (ISA 7.12.2, 16-bit A 16x32):
      // lane<16: row=rsel, K = 32s+{0..7} then 32s+{16..23}
      // lane>=16: row=rsel, K = 32s+{8..15} then 32s+{24..31}
      const _Float16* ap = featS + (rowbase + rsel) * INP + 32 * s + hi * 8;
      v16h a;
#pragma unroll
      for (int e = 0; e < 8; ++e) a[e] = ap[e];
#pragma unroll
      for (int e = 0; e < 8; ++e) a[8 + e] = ap[16 + e];
#pragma unroll
      for (int ct = 0; ct < 8; ++ct) {
        const v16h bw = w1v[(s * 8 + ct) * 32 + lane];
        acc[ct] = __builtin_amdgcn_wmma_f32_16x16x32_f16(
            false, a, false, bw, (short)0, acc[ct], false, false);
      }
    }

    // bias + tanh, H tile to LDS (f16, 16 x 128)
    _Float16* h = hS[w];
#pragma unroll
    for (int ct = 0; ct < 8; ++ct) {
      const int col = 16 * ct + rsel;
      const float bias = b1[col];
#pragma unroll
      for (int v = 0; v < 8; ++v) {
        const int row = v + 8 * hi;
        h[row * HID + col] = (_Float16)tanhf(acc[ct][v] + bias);
      }
    }
    __syncthreads();

    // ---- layer 2: (16 x 128) x (128 x 128), 32 WMMAs ----
    v8f acc2[8];
#pragma unroll
    for (int ct = 0; ct < 8; ++ct) acc2[ct] = zero;
#pragma unroll
    for (int s = 0; s < 4; ++s) {
      const _Float16* ap = h + rsel * HID + 32 * s + hi * 8;
      v16h a;
#pragma unroll
      for (int e = 0; e < 8; ++e) a[e] = ap[e];
#pragma unroll
      for (int e = 0; e < 8; ++e) a[8 + e] = ap[16 + e];
#pragma unroll
      for (int ct = 0; ct < 8; ++ct) {
        const v16h bw = w2v[(s * 8 + ct) * 32 + lane];
        acc2[ct] = __builtin_amdgcn_wmma_f32_16x16x32_f16(
            false, a, false, bw, (short)0, acc2[ct], false, false);
      }
    }

    // bias + tanh + max-pool over K via native LDS atomics
#pragma unroll
    for (int ct = 0; ct < 8; ++ct) {
      const int col = 16 * ct + rsel;
      const float bias = b2[col];
#pragma unroll
      for (int v = 0; v < 8; ++v) {
        const int row = rowbase + v + 8 * hi;     // block-local row 0..159
        const float x = tanhf(acc2[ct][v] + bias);
        atomicMaxF(&pooledS[(row / KK) * HID + col], x);
      }
    }
    __syncthreads();
  }

  // ---- output layer: energy = pooled . w3 + b3 ----
  if (tid < PPB) {
    float acc = b3[0];
    for (int c = 0; c < HID; ++c)
      acc += __int_as_float(pooledS[tid * HID + c]) * w3[c];
    out[blockIdx.x * PPB + tid] = acc;
  }
}

// ---------------------------------------------------------------------------
extern "C" void kernel_launch(void* const* d_in, const int* in_sizes, int n_in,
                              void* d_out, int out_size, void* d_ws, size_t ws_size,
                              hipStream_t stream) {
  (void)in_sizes; (void)n_in; (void)out_size; (void)ws_size;
  const float* position = (const float*)d_in[0];
  const float* orient   = (const float*)d_in[1];
  const float* w1       = (const float*)d_in[2];
  const float* b1       = (const float*)d_in[3];
  const float* w2       = (const float*)d_in[4];
  const float* b2       = (const float*)d_in[5];
  const float* w3       = (const float*)d_in[6];
  const float* b3       = (const float*)d_in[7];
  float* out = (float*)d_out;

  // workspace layout
  const size_t NBR_BYTES = (size_t)BB * NP * KK * sizeof(int);      // 1,310,720 (32B aligned)
  int* nbr = (int*)d_ws;
  _Float16* w1f = (_Float16*)((char*)d_ws + NBR_BYTES);             // 3*8*32*16 halves
  _Float16* w2f = w1f + 3 * 8 * 32 * 16;                            // 4*8*32*16 halves

  prep_weights<<<1, 256, 0, stream>>>(w1, w2, w1f, w2f);
  topk_kernel<<<BB * (NP / 256), 256, 0, stream>>>(position, nbr);
  fused_mlp<<<(BB * NP) / PPB, THREADS_F, 0, stream>>>(
      position, orient, nbr, w1f, b1, w2f, b2, w3, b3, out);
}
